// PairWeightedSelfAttention_33844342292966
// MI455X (gfx1250) — matrix operand order repaired and checked by reading, not compiled, not run on hardware
//
#include <hip/hip_runtime.h>
#include <hip/hip_bf16.h>

// Sizes fixed by the reference: T=512, B=2, E=512, H=8, D=64, P=128, S=T.
#define TT 512
#define BB 2
#define EE 512
#define HH 8
#define DD 64
#define PP 128

typedef __attribute__((ext_vector_type(16))) __bf16 v16bf;
typedef __attribute__((ext_vector_type(8)))  __bf16 v8bf;
typedef __attribute__((ext_vector_type(8)))  float  v8f;
typedef __attribute__((ext_vector_type(4)))  float  v4f;

// ---------------------------------------------------------------------------
// WMMA helpers (bf16 16x16x32, f32 accumulate). Layouts used (A/B: this file's k order; C per ISA 7.12.2):
//  A (16x32): lane L<16 -> row M=L, K = 0..15 contiguous; lane L+16 -> K = 16..31
//  B (32x16): lane L<16 -> col N=L, K = 0..15 contiguous; lane L+16 -> K = 16..31
//  C (16x16): VGPR j, lanes 0-15 -> M=j, lanes 16-31 -> M=j+8; N = lane&15
// ---------------------------------------------------------------------------
__device__ inline v16bf cat8(v8bf lo, v8bf up) {
    return __builtin_shufflevector(lo, up, 0,1,2,3,4,5,6,7,8,9,10,11,12,13,14,15);
}

// A fragment from bf16 row-major source; base = row start + k-base. hi = lane>>4.
__device__ inline v16bf loadA_bf16(const __bf16* base, int hi) {
    v8bf lo = *(const v8bf*)(base + hi * 16);
    v8bf up = *(const v8bf*)(base + hi * 16 + 8);
    return cat8(lo, up);
}
// B fragment from bf16 source where column n is contiguous along K.
__device__ inline v16bf loadB_bf16(const __bf16* base, int hi) {
    v8bf lo = *(const v8bf*)(base + hi * 16);
    v8bf up = *(const v8bf*)(base + hi * 16 + 8);
    return cat8(lo, up);
}
__device__ inline v16bf cvt16(v4f x0, v4f x1, v4f y0, v4f y1) {
    v16bf r;
#pragma unroll
    for (int i = 0; i < 4; ++i) {
        r[i]      = (__bf16)x0[i];
        r[4 + i]  = (__bf16)x1[i];
        r[8 + i]  = (__bf16)y0[i];
        r[12 + i] = (__bf16)y1[i];
    }
    return r;
}
// A fragment from fp32 source (converted to bf16 in-register).
__device__ inline v16bf loadA_f32(const float* base, int hi) {
    const v4f* p0 = (const v4f*)(base + hi * 16);
    const v4f* p1 = (const v4f*)(base + hi * 16 + 8);
    return cvt16(p0[0], p0[1], p1[0], p1[1]);
}
// Same but with non-temporal hint (TH=NT): for single-use streams bigger than L2.
__device__ inline v16bf loadA_f32_nt(const float* base, int hi) {
    const v4f* p0 = (const v4f*)(base + hi * 16);
    const v4f* p1 = (const v4f*)(base + hi * 16 + 8);
    v4f x0 = __builtin_nontemporal_load(p0);
    v4f x1 = __builtin_nontemporal_load(p0 + 1);
    v4f y0 = __builtin_nontemporal_load(p1);
    v4f y1 = __builtin_nontemporal_load(p1 + 1);
    return cvt16(x0, x1, y0, y1);
}
// B fragment from fp32 source, column contiguous along K.
__device__ inline v16bf loadB_f32(const float* base, int hi) {
    const v4f* p = (const v4f*)(base + hi * 16);
    return cvt16(p[0], p[1], p[2], p[3]);
}

__device__ inline v8f wmma_bf16(v16bf a, v16bf b, v8f c) {
    return __builtin_amdgcn_wmma_f32_16x16x32_bf16(false, a, false, b, (short)0, c,
                                                   false, false);
}

// ---------------------------------------------------------------------------
// Kernel 1: q/k/v projections.  rows r = t*B+b (1024), cols e (512), K = 512.
//   q (scaled by D^-0.5) -> qbf[r][e]  (bf16)
//   k                    -> kbf[r][e]  (bf16)
//   v                    -> vT[b][h][d][s] (bf16, transposed for AV B-operand)
// ---------------------------------------------------------------------------
__global__ __launch_bounds__(256) void k_proj_qkv(
    const float* __restrict__ query, const float* __restrict__ wq,
    const float* __restrict__ wk, const float* __restrict__ wv,
    __bf16* __restrict__ qbf, __bf16* __restrict__ kbf, __bf16* __restrict__ vT)
{
    const int lane = threadIdx.x & 31, wave = threadIdx.x >> 5;
    const int hi = lane >> 4, ln = lane & 15;
    const int mbase = blockIdx.x * 16;
    const int nbase = blockIdx.y * 128 + wave * 16;
    const int z = blockIdx.z;
    const float* W = (z == 0) ? wq : (z == 1) ? wk : wv;

    const float* arow = query + (size_t)(mbase + ln) * EE;
    const float* brow = W + (size_t)(nbase + ln) * EE;   // col e of W^T = row e of W
    v8f acc = 0.0f;
    for (int kk = 0; kk < EE; kk += 32)
        acc = wmma_bf16(loadA_f32(arow + kk, hi), loadB_f32(brow + kk, hi), acc);

    const int e = nbase + ln;
#pragma unroll
    for (int j = 0; j < 8; ++j) {
        int r = mbase + j + 8 * hi;
        float val = acc[j];
        if (z == 0) {
            qbf[(size_t)r * EE + e] = (__bf16)(val * 0.125f);   // D^-0.5
        } else if (z == 1) {
            kbf[(size_t)r * EE + e] = (__bf16)val;
        } else {
            int s = r >> 1, b = r & 1, h = e >> 6, d = e & 63;
            vT[(((size_t)b * HH + h) * DD + d) * TT + s] = (__bf16)val;
        }
    }
}

// ---------------------------------------------------------------------------
// Kernel 2: qp[r][h][p] = sum_d qbf[r][h*64+d] * pair_w[h*64+d][p]   (bf16 out)
// ---------------------------------------------------------------------------
__global__ __launch_bounds__(256) void k_proj_qp(
    const __bf16* __restrict__ qbf, const float* __restrict__ pair_w,
    __bf16* __restrict__ qp)
{
    const int lane = threadIdx.x & 31, wave = threadIdx.x >> 5;
    const int hi = lane >> 4, ln = lane & 15;
    const int mbase = blockIdx.x * 16;
    const int h = blockIdx.y;
    const int pbase = wave * 16;

    const __bf16* arow = qbf + (size_t)(mbase + ln) * EE + h * DD;
    v8f acc = 0.0f;
    for (int kk = 0; kk < DD; kk += 32) {
        v16bf a = loadA_bf16(arow + kk, hi);
        v16bf b;
#pragma unroll
        for (int i = 0; i < 16; ++i) {   // pair_w gathered along d (stride P)
            int d = kk + hi * 16 + i;
            b[i] = (__bf16)pair_w[(size_t)(h * DD + d) * PP + (pbase + ln)];
        }
        acc = wmma_bf16(a, b, acc);
    }
    const int p = pbase + ln;
#pragma unroll
    for (int j = 0; j < 8; ++j) {
        int r = mbase + j + 8 * hi;
        qp[((size_t)r * HH + h) * PP + p] = (__bf16)acc[j];
    }
}

// ---------------------------------------------------------------------------
// Kernel 3: content logits: logits0[bh][t][s] = q_h(TxD) @ k_h^T   (fp32 out)
// ---------------------------------------------------------------------------
__global__ __launch_bounds__(256) void k_logits0(
    const __bf16* __restrict__ qbf, const __bf16* __restrict__ kbf,
    float* __restrict__ logits0)
{
    const int lane = threadIdx.x & 31, wave = threadIdx.x >> 5;
    const int hi = lane >> 4, ln = lane & 15;
    const int bh = blockIdx.z, b = bh >> 3, h = bh & 7;
    const int tbase = blockIdx.x * 16;
    const int sbase = blockIdx.y * 128 + wave * 16;

    const __bf16* arow = qbf + ((size_t)(tbase + ln) * BB + b) * EE + h * DD;
    const __bf16* brow = kbf + ((size_t)(sbase + ln) * BB + b) * EE + h * DD;
    v8f acc = 0.0f;
    for (int kk = 0; kk < DD; kk += 32)
        acc = wmma_bf16(loadA_bf16(arow + kk, hi), loadB_bf16(brow + kk, hi), acc);

    const int s = sbase + ln;
#pragma unroll
    for (int j = 0; j < 8; ++j) {
        int t = tbase + j + 8 * hi;
        logits0[((size_t)bh * TT + t) * TT + s] = acc[j];
    }
}

// ---------------------------------------------------------------------------
// Kernel 4 (hot, bandwidth-bound on `pairs`): one workgroup per (b,t).
//   bias(512 x 8) = pairs[b,t] (512x128, fp32->bf16, NT stream) @ qp^T (128x8)
//   logits = bias + logits0 ; per-head softmax in LDS ; bf16 weights out.
// ---------------------------------------------------------------------------
__global__ __launch_bounds__(256) void k_bias_softmax(
    const float* __restrict__ pairs, const __bf16* __restrict__ qp,
    const float* __restrict__ logits0, __bf16* __restrict__ wbuf)
{
    __shared__ float lgt[HH * TT];   // 16 KB
    const int lane = threadIdx.x & 31, wave = threadIdx.x >> 5;
    const int hi = lane >> 4, ln = lane & 15;
    const int bt = blockIdx.x, b = bt & 1, t = bt >> 1;

    // Preload the 4 qp B-fragments (K = 128). Column n = head; heads 8..15 -> 0.
    v16bf bq[4];
    if (ln < HH) {
        const __bf16* qrow = qp + ((size_t)(t * BB + b) * HH + ln) * PP;
#pragma unroll
        for (int q4 = 0; q4 < 4; ++q4) bq[q4] = loadB_bf16(qrow + q4 * 32, hi);
    } else {
#pragma unroll
        for (int q4 = 0; q4 < 4; ++q4)
#pragma unroll
            for (int i = 0; i < 16; ++i) bq[q4][i] = (__bf16)0.0f;
    }

    const float* pslab = pairs + ((size_t)b * TT + t) * TT * PP;
    for (int it = 0; it < 4; ++it) {
        const int sbase = (wave + it * 8) * 16;
        const float* arow = pslab + (size_t)(sbase + ln) * PP;
        v8f acc = 0.0f;
#pragma unroll
        for (int q4 = 0; q4 < 4; ++q4)
            acc = wmma_bf16(loadA_f32_nt(arow + q4 * 32, hi), bq[q4], acc);

        const int h = ln;                        // C tile: N = head
        if (h < HH) {
            const float* l0 = logits0 + ((size_t)(b * HH + h) * TT + t) * TT;
#pragma unroll
            for (int j = 0; j < 8; ++j) {
                int s = sbase + j + 8 * hi;
                lgt[h * TT + s] = acc[j] + l0[s];
            }
        }
    }
    __syncthreads();

    // Softmax over s: one wave32 per head, 16 values per lane.
    {
        const int h = wave;
        float v[16];
        float mx = -3.0e38f;
#pragma unroll
        for (int i = 0; i < 16; ++i) {
            v[i] = lgt[h * TT + lane + i * 32];
            mx = fmaxf(mx, v[i]);
        }
#pragma unroll
        for (int off = 16; off >= 1; off >>= 1)
            mx = fmaxf(mx, __shfl_xor(mx, off, 32));
        float sum = 0.0f;
#pragma unroll
        for (int i = 0; i < 16; ++i) { v[i] = __expf(v[i] - mx); sum += v[i]; }
#pragma unroll
        for (int off = 16; off >= 1; off >>= 1)
            sum += __shfl_xor(sum, off, 32);
        const float inv = 1.0f / sum;
        __bf16* wrow = wbuf + ((size_t)(b * HH + h) * TT + t) * TT;
#pragma unroll
        for (int i = 0; i < 16; ++i)
            wrow[lane + i * 32] = (__bf16)(v[i] * inv);
    }
}

// ---------------------------------------------------------------------------
// Kernel 5: out_h(TxD) = w(TxS) @ v_h(SxD); write bf16 ao[r=t*B+b][e=h*64+d].
// ---------------------------------------------------------------------------
__global__ __launch_bounds__(256) void k_av(
    const __bf16* __restrict__ wbuf, const __bf16* __restrict__ vT,
    __bf16* __restrict__ ao)
{
    const int lane = threadIdx.x & 31, wave = threadIdx.x >> 5;
    const int hi = lane >> 4, ln = lane & 15;
    const int bh = blockIdx.z, b = bh >> 3, h = bh & 7;
    const int mbase = blockIdx.x * 32 + (wave & 1) * 16;   // t tile
    const int nbase = (wave >> 1) * 16;                    // d tile (D=64 -> 4 tiles)

    const __bf16* arow = wbuf + ((size_t)bh * TT + (mbase + ln)) * TT;
    const __bf16* brow = vT + ((size_t)bh * DD + (nbase + ln)) * TT;
    v8f acc = 0.0f;
    for (int kk = 0; kk < TT; kk += 32)
        acc = wmma_bf16(loadA_bf16(arow + kk, hi), loadB_bf16(brow + kk, hi), acc);

    const int e = h * DD + nbase + ln;
#pragma unroll
    for (int j = 0; j < 8; ++j) {
        int tloc = mbase + j + 8 * hi;
        int r = tloc * BB + b;
        ao[(size_t)r * EE + e] = (__bf16)acc[j];
    }
}

// ---------------------------------------------------------------------------
// Kernel 6: final = ao @ out_w^T + out_b  -> fp32 (T,B,E)
// ---------------------------------------------------------------------------
__global__ __launch_bounds__(256) void k_final(
    const __bf16* __restrict__ ao, const float* __restrict__ out_w,
    const float* __restrict__ out_b, float* __restrict__ out)
{
    const int lane = threadIdx.x & 31, wave = threadIdx.x >> 5;
    const int hi = lane >> 4, ln = lane & 15;
    const int mbase = blockIdx.x * 16;
    const int nbase = blockIdx.y * 128 + wave * 16;

    const __bf16* arow = ao + (size_t)(mbase + ln) * EE;
    const float* brow = out_w + (size_t)(nbase + ln) * EE;
    v8f acc = 0.0f;
    for (int kk = 0; kk < EE; kk += 32)
        acc = wmma_bf16(loadA_bf16(arow + kk, hi), loadB_f32(brow + kk, hi), acc);

    const int e = nbase + ln;
    const float bias = out_b[e];
#pragma unroll
    for (int j = 0; j < 8; ++j) {
        int r = mbase + j + 8 * hi;
        out[(size_t)r * EE + e] = acc[j] + bias;
    }
}

// ---------------------------------------------------------------------------
extern "C" void kernel_launch(void* const* d_in, const int* in_sizes, int n_in,
                              void* d_out, int out_size, void* d_ws, size_t ws_size,
                              hipStream_t stream) {
    const float* query  = (const float*)d_in[0];
    const float* pairs  = (const float*)d_in[1];
    const float* q_w    = (const float*)d_in[2];
    const float* k_w    = (const float*)d_in[3];
    const float* v_w    = (const float*)d_in[4];
    const float* out_w  = (const float*)d_in[5];
    const float* out_b  = (const float*)d_in[6];
    const float* pair_w = (const float*)d_in[7];
    float* out = (float*)d_out;

    char* ws = (char*)d_ws;
    __bf16* qbf    = (__bf16*)(ws);                        // 1 MB (1024x512 bf16)
    __bf16* kbf    = (__bf16*)(ws + (1u << 20));           // 1 MB
    __bf16* vT     = (__bf16*)(ws + (2u << 20));           // 1 MB (B,H,D,S)
    __bf16* qp     = (__bf16*)(ws + (3u << 20));           // 2 MB (1024,H,P)
    float*  logit0 = (float*)(ws + (5u << 20));            // 16 MB (B,H,T,S) f32
    __bf16* wbuf   = (__bf16*)(ws + (13u << 20));          // 8 MB (B,H,T,S) bf16
    __bf16* ao     = (__bf16*)(ws + (17u << 20));          // 1 MB (1024x512)

    dim3 blk(256);
    k_proj_qkv   <<<dim3(64, 4, 3),  blk, 0, stream>>>(query, q_w, k_w, v_w, qbf, kbf, vT);
    k_proj_qp    <<<dim3(64, 8),     blk, 0, stream>>>(qbf, pair_w, qp);
    k_logits0    <<<dim3(32, 4, 16), blk, 0, stream>>>(qbf, kbf, logit0);
    k_bias_softmax<<<dim3(1024),     blk, 0, stream>>>(pairs, qp, logit0, wbuf);
    k_av         <<<dim3(16, 1, 16), blk, 0, stream>>>(wbuf, vT, ao);
    k_final      <<<dim3(64, 4),     blk, 0, stream>>>(ao, out_w, out_b, out);
}
